// CascadedMartingaleEncoder_2199023256195
// MI455X (gfx1250) — matrix operand order repaired and checked by reading, not compiled, not run on hardware
//
#include <hip/hip_runtime.h>
#include <math.h>
#include <stdint.h>

// ---------------------------------------------------------------------------
// CDNA5 (gfx1250) implementation. wave32, WMMA f32<=f16 16x16x32, async-to-LDS
// weight staging (ASYNCcnt), f16 pre-converted zero-padded weights so the GEMM
// hot loop has no bounds checks.
// ---------------------------------------------------------------------------

typedef __attribute__((ext_vector_type(16))) _Float16 v16h;
typedef __attribute__((ext_vector_type(8)))  _Float16 v8h;
typedef __attribute__((ext_vector_type(4)))  _Float16 v4h;
typedef __attribute__((ext_vector_type(8)))  float    v8f;

#define EPSC 1e-6f

// ---- WMMA fragment loaders (per CDNA5 ISA 7.12.2 layouts) ------------------
// A-matrix 16x32 f16: lane L row = L%16; VGPR v pair K = 2*(v%4)+p + 8*hi + 16*(v/4)
__device__ __forceinline__ v16h load_fragA(const _Float16* rowp, int hi) {
  v16h f;
#pragma unroll
  for (int v = 0; v < 8; ++v) {
    int k = 2 * (v & 3) + (hi ? 8 : 0) + ((v >> 2) ? 16 : 0);
    f[2 * v]     = rowp[k];
    f[2 * v + 1] = rowp[k + 1];
  }
  return f;
}
// B-matrix 32x16 f16 (stored as B^T rows [n][k]): lane L col n = L%16;
// element e: K = 8*hi + e (e<8), 16 + 8*hi + (e-8) (e>=8), matching A.
__device__ __forceinline__ v16h load_fragB(const _Float16* rowp, int hi) {
  v16h f;
  const _Float16* p = rowp + (hi ? 8 : 0);
#pragma unroll
  for (int e = 0; e < 16; ++e) f[e] = p[e + (e & 8)];
  return f;
}

__device__ __forceinline__ v8f wmma_f16(v16h a, v16h b, v8f c) {
  return __builtin_amdgcn_wmma_f32_16x16x32_f16(false, a, false, b, (short)0, c,
                                                false, false);
}

__device__ __forceinline__ float redmax16(float t) {
#pragma unroll
  for (int off = 8; off; off >>= 1) t = fmaxf(t, __shfl_xor(t, off, 16));
  return t;
}
__device__ __forceinline__ float redsum16(float t) {
#pragma unroll
  for (int off = 8; off; off >>= 1) t += __shfl_xor(t, off, 16);
  return t;
}

// async copy 16 bytes global(f16) -> LDS, tracked by ASYNCcnt
__device__ __forceinline__ void async_copy_b128(const _Float16* g, _Float16* l) {
  unsigned lds_off = (unsigned)(uintptr_t)l;  // low 32 bits = LDS byte offset
  asm volatile("global_load_async_to_lds_b128 %0, %1, off"
               :: "v"(lds_off), "v"(g) : "memory");
}
__device__ __forceinline__ void wait_asynccnt0() {
  asm volatile("s_wait_asynccnt 0x0" ::: "memory");
}

// ---------------------------------------------------------------------------
// K0: convert weight matrix [N,K] f32 -> [N,Kp] f16, zero-padded K..Kp.
// ---------------------------------------------------------------------------
__global__ void cvt_weights(const float* __restrict__ W,
                            _Float16* __restrict__ Wh, int N, int K, int Kp) {
  int idx = blockIdx.x * blockDim.x + threadIdx.x;
  if (idx >= N * Kp) return;
  int n = idx / Kp, k = idx - n * Kp;
  Wh[idx] = (k < K) ? (_Float16)W[(size_t)n * K + k] : (_Float16)0.f;
}

// ---------------------------------------------------------------------------
// K1: texture martingale stats fused with 1024->12 reduce conv.
// ---------------------------------------------------------------------------
__global__ __launch_bounds__(256) void martingale_reduce(
    const float* __restrict__ x, const float* __restrict__ rw,
    const float* __restrict__ rb, float* __restrict__ m12) {
  const int pix = blockIdx.x * 256 + threadIdx.x;  // 0..16383
  const int b = pix >> 12;
  const int hw = pix & 4095;
  const int hh = hw >> 6, ww = hw & 63;
  const bool interior = (hh >= 1) & (hh < 63) & (ww >= 1) & (ww < 63);

  float acc[12];
#pragma unroll
  for (int o = 0; o < 12; ++o) acc[o] = 0.f;

  for (int c = 0; c < 256; ++c) {
    const float* xc = x + ((size_t)(b * 256 + c) << 12);
    float p[9];
    if (interior) {
      const float* base = xc + (hh - 1) * 64 + (ww - 1);
#pragma unroll
      for (int di = 0; di < 3; ++di)
#pragma unroll
        for (int dj = 0; dj < 3; ++dj) p[di * 3 + dj] = base[di * 64 + dj];
    } else {
      int n = 0;
#pragma unroll
      for (int di = -1; di <= 1; ++di)
#pragma unroll
        for (int dj = -1; dj <= 1; ++dj) {
          int y = hh + di, xx = ww + dj;
          p[n++] =
              (y >= 0 && y < 64 && xx >= 0 && xx < 64) ? xc[y * 64 + xx] : 0.f;
        }
    }
    float mean = 0.f;
#pragma unroll
    for (int i = 0; i < 9; ++i) mean += p[i];
    mean *= (1.f / 9.f);
    float var = 0.f, aabs = 0.f, en = 0.f, ent = 0.f;
#pragma unroll
    for (int i = 0; i < 9; ++i) {
      float cc = p[i] - mean;
      var += cc * cc;
      aabs += fabsf(cc);
      en += p[i] * p[i];
      ent += p[i] * __logf(p[i] + EPSC);
    }
    float sd = sqrtf(var * (1.f / 8.f)) + EPSC;   // unbiased std + eps
    float f0 = var / (9.f * sd * sd);             // contrast
    float f1 = en * (1.f / 9.f);                  // energy
    float f2 = -ent * (1.f / 9.f);                // entropy
    float f3 = 1.f / (1.f + aabs * (1.f / 9.f));  // homogeneity
    // THETA=1: M = exp(log(f+eps) - 0.5) = (f+eps)*e^-0.5
    const float e_half = 0.60653065971f;
    float M0 = (f0 + EPSC) * e_half, M1 = (f1 + EPSC) * e_half;
    float M2 = (f2 + EPSC) * e_half, M3 = (f3 + EPSC) * e_half;
    const float* w0 = rw + (c << 2);  // wave-uniform -> scalar loads
#pragma unroll
    for (int o = 0; o < 12; ++o) {
      const float* wr = w0 + o * 1024;
      acc[o] += M0 * wr[0] + M1 * wr[1] + M2 * wr[2] + M3 * wr[3];
    }
  }
#pragma unroll
  for (int o = 0; o < 12; ++o)
    m12[((size_t)(b * 12 + o) << 12) + hw] = acc[o] + rb[o];
}

// ---------------------------------------------------------------------------
// K2: build fuse-GEMM A matrix [4096, 524] = [x_next pixel-major | m_up(2x avg)]
// ---------------------------------------------------------------------------
__global__ void pack_fuse(const float* __restrict__ xn,
                          const float* __restrict__ m12,
                          float* __restrict__ Af) {
  int idx = blockIdx.x * blockDim.x + threadIdx.x;
  if (idx >= 4096 * 524) return;
  int p = idx / 524, c = idx - p * 524;
  int b = p >> 10, hw = p & 1023;
  float v;
  if (c < 512) {
    v = xn[((size_t)(b * 512 + c) << 10) + hw];
  } else {
    int cc = c - 512;
    int ho = hw >> 5, wo = hw & 31;
    const float* mp = m12 + ((size_t)(b * 12 + cc) << 12);
    int h2 = ho * 2, w2 = wo * 2;
    v = 0.25f * (mp[h2 * 64 + w2] + mp[h2 * 64 + w2 + 1] +
                 mp[(h2 + 1) * 64 + w2] + mp[(h2 + 1) * 64 + w2 + 1]);
  }
  Af[idx] = v;
}

// ---------------------------------------------------------------------------
// WMMA GEMM: C[M,N] = A[M,K] * Wh[N,Kp]^T + bias[N]
// Wh is f16, K zero-padded to Kp (mult of 32) -> NO bounds checks anywhere.
// (A may be over-read in its k-tail; those lanes hit Wh's zero pad.)
// Block 256 thr = 8 waves, tile 32(M) x 128(N), K step 32.
// B tiles staged via global_load_async_to_lds_b128 (ASYNCcnt).
// ---------------------------------------------------------------------------
__global__ __launch_bounds__(256) void gemm_bt_wmma(
    const float* __restrict__ A, int lda, const _Float16* __restrict__ Wh,
    int ldw, const float* __restrict__ bias, float* __restrict__ C, int ldc,
    int M, int N, int Kp) {
  __shared__ _Float16 As[32][32];
  __shared__ _Float16 Bs[128][32];
  const int tid = threadIdx.x;
  const int lane = tid & 31, wave = tid >> 5, hi = lane >> 4;
  const int bm = blockIdx.y * 32, bn = blockIdx.x * 128;
  const int tr = wave >> 2, tc0 = (wave & 3) * 2;
  const int arow = 16 * tr + (lane & 15);

  const int ar = tid >> 3, akk = (tid & 7) * 4;        // A stage coords
  const int bnn = tid >> 1, bkk = (tid & 1) * 16;      // B stage coords
  const float* Ag = A + (size_t)(bm + ar) * lda + akk;
  const _Float16* Bg = Wh + (size_t)(bn + bnn) * ldw + bkk;

  v8f c0 = {}, c1 = {};

  for (int k0 = 0; k0 < Kp; k0 += 32) {
    __syncthreads();
    // B tile: pure f16 copy global -> LDS via async engine (2 x 16B)
    async_copy_b128(Bg + k0, &Bs[bnn][bkk]);
    async_copy_b128(Bg + k0 + 8, &Bs[bnn][bkk + 8]);
    // A tile: float4 load + pack to f16 (8B ds store)
    {
      float4 va = *(const float4*)(Ag + k0);
      v4h h;
      h[0] = (_Float16)va.x; h[1] = (_Float16)va.y;
      h[2] = (_Float16)va.z; h[3] = (_Float16)va.w;
      *(v4h*)&As[ar][akk] = h;
    }
    if (k0 + 32 < Kp)
      __builtin_prefetch(Ag + k0 + 32, 0, 1);
    wait_asynccnt0();
    __syncthreads();
    v16h a  = load_fragA(&As[arow][0], hi);
    v16h b0 = load_fragB(&Bs[16 * tc0 + (lane & 15)][0], hi);
    v16h b1 = load_fragB(&Bs[16 * (tc0 + 1) + (lane & 15)][0], hi);
    c0 = wmma_f16(a, b0, c0);
    c1 = wmma_f16(a, b1, c1);
  }
#pragma unroll
  for (int r = 0; r < 8; ++r) {
    int row = bm + 16 * tr + r + 8 * hi;
    int col0 = bn + 16 * tc0 + (lane & 15);
    C[(size_t)row * ldc + col0]      = c0[r] + bias[col0];
    C[(size_t)row * ldc + col0 + 16] = c1[r] + bias[col0 + 16];
  }
}

// ---------------------------------------------------------------------------
// K4: qk = LayerNorm(x_flat + biasbuf) * g + b, rows of 512
// ---------------------------------------------------------------------------
__global__ __launch_bounds__(256) void add_layernorm(
    const float* __restrict__ xf, const float* __restrict__ bb,
    const float* __restrict__ g, const float* __restrict__ be,
    float* __restrict__ out) {
  __shared__ float red[256];
  const int row = blockIdx.x, t = threadIdx.x;
  const float* xr = xf + (size_t)row * 512;
  const float* br = bb + (size_t)row * 512;
  float v0 = xr[t] + br[t], v1 = xr[t + 256] + br[t + 256];
  red[t] = v0 + v1;
  __syncthreads();
  for (int s = 128; s; s >>= 1) {
    if (t < s) red[t] += red[t + s];
    __syncthreads();
  }
  float mu = red[0] * (1.f / 512.f);
  __syncthreads();
  float d0 = v0 - mu, d1 = v1 - mu;
  red[t] = d0 * d0 + d1 * d1;
  __syncthreads();
  for (int s = 128; s; s >>= 1) {
    if (t < s) red[t] += red[t + s];
    __syncthreads();
  }
  float rstd = rsqrtf(red[0] * (1.f / 512.f) + 1e-5f);
  out[(size_t)row * 512 + t]       = d0 * rstd * g[t] + be[t];
  out[(size_t)row * 512 + t + 256] = d1 * rstd * g[t + 256] + be[t + 256];
}

// ---------------------------------------------------------------------------
// K5: flash attention, dh=64, 8 heads, S=1024.
// qkbuf:[B*S,1024] = [Q|K], vbuf:[B*S,512]. Wave owns 16 q rows; k tiles of 32.
// ---------------------------------------------------------------------------
__global__ __launch_bounds__(256) void flash_attn(
    const float* __restrict__ qkbuf, const float* __restrict__ vbuf,
    float* __restrict__ obuf, int S) {
  __shared__ _Float16 Ks[32][64];     // k rows (B^T for scores)
  __shared__ _Float16 Vt[64][32];     // V transposed (B^T for P*V)
  __shared__ _Float16 Ps[8][16][32];  // per-wave probabilities (A for P*V)

  const int tid = threadIdx.x, lane = tid & 31, wave = tid >> 5, hi = lane >> 4;
  const int b = blockIdx.z, h = blockIdx.y, qb = blockIdx.x;
  const size_t rowbase = (size_t)b * S;
  const int qcol = h * 64;
  const size_t qrow = rowbase + qb * 128 + wave * 16 + (lane & 15);
  const int si = tid >> 3, sd = (tid & 7) * 8;  // staging coords

  // Q A-fragments (pre-scaled by 1/sqrt(dh))
  v16h aq0, aq1;
#pragma unroll
  for (int v = 0; v < 8; ++v) {
    int k = 2 * (v & 3) + (hi ? 8 : 0) + ((v >> 2) ? 16 : 0);
    const float* qp = qkbuf + qrow * 1024 + qcol;
    aq0[2 * v]     = (_Float16)(0.125f * qp[k]);
    aq0[2 * v + 1] = (_Float16)(0.125f * qp[k + 1]);
    aq1[2 * v]     = (_Float16)(0.125f * qp[32 + k]);
    aq1[2 * v + 1] = (_Float16)(0.125f * qp[32 + k + 1]);
  }

  float m_run[8], l_run[8];
  v8f o0 = {}, o1 = {}, o2 = {}, o3 = {};
#pragma unroll
  for (int r = 0; r < 8; ++r) { m_run[r] = -1e30f; l_run[r] = 0.f; }

  for (int kt = 0; kt < S; kt += 32) {
    __syncthreads();
    {  // stage K (row-major) and V (transposed), vector loads
      size_t gr = rowbase + kt + si;
      const float* kp = qkbuf + gr * 1024 + 512 + qcol + sd;
      const float* vp = vbuf + gr * 512 + qcol + sd;
      float4 ka = *(const float4*)kp, kb = *(const float4*)(kp + 4);
      float4 va = *(const float4*)vp, vb = *(const float4*)(vp + 4);
      v8h hk;
      hk[0] = (_Float16)ka.x; hk[1] = (_Float16)ka.y;
      hk[2] = (_Float16)ka.z; hk[3] = (_Float16)ka.w;
      hk[4] = (_Float16)kb.x; hk[5] = (_Float16)kb.y;
      hk[6] = (_Float16)kb.z; hk[7] = (_Float16)kb.w;
      *(v8h*)&Ks[si][sd] = hk;
      _Float16 hv[8];
      hv[0] = (_Float16)va.x; hv[1] = (_Float16)va.y;
      hv[2] = (_Float16)va.z; hv[3] = (_Float16)va.w;
      hv[4] = (_Float16)vb.x; hv[5] = (_Float16)vb.y;
      hv[6] = (_Float16)vb.z; hv[7] = (_Float16)vb.w;
#pragma unroll
      for (int j = 0; j < 8; ++j) Vt[sd + j][si] = hv[j];
    }
    __syncthreads();

    // scores: two 16x16 tiles (k cols 0..15, 16..31), K-dim dh=64 -> 2 chunks
    v8f s0 = {}, s1 = {};
    s0 = wmma_f16(aq0, load_fragB(&Ks[(lane & 15)][0], hi), s0);
    s0 = wmma_f16(aq1, load_fragB(&Ks[(lane & 15)][32], hi), s0);
    s1 = wmma_f16(aq0, load_fragB(&Ks[16 + (lane & 15)][0], hi), s1);
    s1 = wmma_f16(aq1, load_fragB(&Ks[16 + (lane & 15)][32], hi), s1);

    // online softmax per row (rows live in 16-lane groups)
#pragma unroll
    for (int r = 0; r < 8; ++r) {
      float mt = redmax16(fmaxf(s0[r], s1[r]));
      float mn = fmaxf(m_run[r], mt);
      float sc = __expf(m_run[r] - mn);
      m_run[r] = mn;
      l_run[r] *= sc;
      o0[r] *= sc; o1[r] *= sc; o2[r] *= sc; o3[r] *= sc;
      float p0 = __expf(s0[r] - mn), p1 = __expf(s1[r] - mn);
      l_run[r] += redsum16(p0 + p1);
      int prow = r + 8 * hi;
      Ps[wave][prow][(lane & 15)]      = (_Float16)p0;
      Ps[wave][prow][16 + (lane & 15)] = (_Float16)p1;
    }
    // wave-private LDS RAW: ensure P stores land before fragment reload
    asm volatile("s_wait_dscnt 0" ::: "memory");

    v16h ap = load_fragA(&Ps[wave][(lane & 15)][0], hi);
    o0 = wmma_f16(ap, load_fragB(&Vt[(lane & 15)][0], hi), o0);
    o1 = wmma_f16(ap, load_fragB(&Vt[16 + (lane & 15)][0], hi), o1);
    o2 = wmma_f16(ap, load_fragB(&Vt[32 + (lane & 15)][0], hi), o2);
    o3 = wmma_f16(ap, load_fragB(&Vt[48 + (lane & 15)][0], hi), o3);
  }

#pragma unroll
  for (int r = 0; r < 8; ++r) {
    float inv = 1.f / l_run[r];
    size_t row = rowbase + qb * 128 + wave * 16 + r + 8 * hi;
    int c = qcol + (lane & 15);
    obuf[row * 512 + c]      = o0[r] * inv;
    obuf[row * 512 + c + 16] = o1[r] * inv;
    obuf[row * 512 + c + 32] = o2[r] * inv;
    obuf[row * 512 + c + 48] = o3[r] * inv;
  }
}

// ---------------------------------------------------------------------------
// K7: [B*S, 512] row-major -> [B,512,32,32] NCHW
// ---------------------------------------------------------------------------
__global__ void transpose_out(const float* __restrict__ y,
                              float* __restrict__ out) {
  int idx = blockIdx.x * blockDim.x + threadIdx.x;
  if (idx >= 4 * 512 * 1024) return;
  int b = idx >> 19;
  int rem = idx & ((1 << 19) - 1);
  int c = rem >> 10, hw = rem & 1023;
  out[idx] = y[((size_t)(b << 10) + hw) * 512 + c];
}

// ---------------------------------------------------------------------------
extern "C" void kernel_launch(void* const* d_in, const int* in_sizes, int n_in,
                              void* d_out, int out_size, void* d_ws,
                              size_t ws_size, hipStream_t stream) {
  (void)in_sizes; (void)n_in; (void)out_size; (void)ws_size;
  const float* x        = (const float*)d_in[0];
  const float* x_next   = (const float*)d_in[1];
  const float* reduce_w = (const float*)d_in[2];
  const float* reduce_b = (const float*)d_in[3];
  const float* fuse_w   = (const float*)d_in[4];
  const float* fuse_b   = (const float*)d_in[5];
  const float* proj_w   = (const float*)d_in[6];
  const float* proj_b   = (const float*)d_in[7];
  const float* ln_g     = (const float*)d_in[8];
  const float* ln_b     = (const float*)d_in[9];
  const float* in_w     = (const float*)d_in[10];
  const float* in_b     = (const float*)d_in[11];
  const float* out_w    = (const float*)d_in[12];
  const float* out_b    = (const float*)d_in[13];
  float* out = (float*)d_out;

  float* ws      = (float*)d_ws;
  float* m12     = ws; ws += 4 * 12 * 64 * 64;   // 196608
  float* Af      = ws; ws += 4096 * 524 + 64;    // +64 pad for k-tail over-read
  float* xflat   = ws; ws += 4096 * 512;
  float* biasbuf = ws; ws += 4096 * 512;
  float* qk      = ws; ws += 4096 * 512;
  float* qkbuf   = ws; ws += 4096 * 1024;
  float* vbuf    = ws; ws += 4096 * 512;
  float* attn    = ws; ws += 4096 * 512;
  float* ybuf    = ws; ws += 4096 * 512;
  // f16 zero-padded weights
  _Float16* hw0   = (_Float16*)ws;
  _Float16* fuseh = hw0;  hw0 += 512 * 544;
  _Float16* projh = hw0;  hw0 += 512 * 32;
  _Float16* inh   = hw0;  hw0 += 1536 * 512;
  _Float16* outh  = hw0;  hw0 += 512 * 512;

  // 0) weight conversion (one-shot, f32 -> zero-padded f16)
  cvt_weights<<<(512 * 544 + 255) / 256, 256, 0, stream>>>(fuse_w, fuseh, 512, 524, 544);
  cvt_weights<<<(512 * 32 + 255) / 256, 256, 0, stream>>>(proj_w, projh, 512, 12, 32);
  cvt_weights<<<(1536 * 512 + 255) / 256, 256, 0, stream>>>(in_w, inh, 1536, 512, 512);
  cvt_weights<<<(512 * 512 + 255) / 256, 256, 0, stream>>>(out_w, outh, 512, 512, 512);

  // 1) texture stats + 1024->12 reduce
  martingale_reduce<<<64, 256, 0, stream>>>(x, reduce_w, reduce_b, m12);
  // 2) downsample + channel pack into [4096, 524]
  pack_fuse<<<(4096 * 524 + 255) / 256, 256, 0, stream>>>(x_next, m12, Af);
  // 3) x_flat = Af @ fuse_w^T + fuse_b          [4096, 512]
  gemm_bt_wmma<<<dim3(512 / 128, 4096 / 32), 256, 0, stream>>>(
      Af, 524, fuseh, 544, fuse_b, xflat, 512, 4096, 512, 544);
  // 4) biasbuf = tm_flat @ proj_w^T + proj_b    [4096, 512] (K=12, padded to 32)
  gemm_bt_wmma<<<dim3(512 / 128, 4096 / 32), 256, 0, stream>>>(
      Af + 512, 524, projh, 32, proj_b, biasbuf, 512, 4096, 512, 32);
  // 5) qk = LN(x_flat + biasbuf)
  add_layernorm<<<4096, 256, 0, stream>>>(xflat, biasbuf, ln_g, ln_b, qk);
  // 6) [Q|K] = qk @ in_w[0:1024]^T + b          [4096, 1024]
  gemm_bt_wmma<<<dim3(1024 / 128, 4096 / 32), 256, 0, stream>>>(
      qk, 512, inh, 512, in_b, qkbuf, 1024, 4096, 1024, 512);
  // 7) V = x_flat @ in_w[1024:1536]^T + b       [4096, 512]
  gemm_bt_wmma<<<dim3(512 / 128, 4096 / 32), 256, 0, stream>>>(
      xflat, 512, inh + 1024 * 512, 512, in_b + 1024, vbuf, 512, 4096, 512,
      512);
  // 8) flash attention (8 heads, dh=64)
  flash_attn<<<dim3(8, 8, 4), 256, 0, stream>>>(qkbuf, vbuf, attn, 1024);
  // 9) y = attn @ out_w^T + out_b               [4096, 512]
  gemm_bt_wmma<<<dim3(512 / 128, 4096 / 32), 256, 0, stream>>>(
      attn, 512, outh, 512, out_b, ybuf, 512, 4096, 512, 512);
  // 10) NCHW transpose
  transpose_out<<<(4 * 512 * 1024 + 255) / 256, 256, 0, stream>>>(ybuf, out);
}
